// AutoEncoder_75282186764667
// MI455X (gfx1250) — matrix operand order repaired and checked by reading, not compiled, not run on hardware
//
#include <hip/hip_runtime.h>
#include <hip/hip_bf16.h>
#include <string.h>

// ---------------- problem constants ----------------
#define Nn   262144          // total elements
#define Bb   32768           // sets
#define Dd   128
#define Hh   256
#define Mm   17
#define R2   (Bb * Mm)       // decoder rows = 557056

// mlp kernel limits
#define MAX_KP 288           // max padded K (enc_main K=273 -> 288)
#define MAX_CP 272           // max padded C (enc_main C=264 -> 272)

typedef __attribute__((ext_vector_type(16))) _Float16 v16h;
typedef __attribute__((ext_vector_type(8)))  _Float16 v8h;
typedef __attribute__((ext_vector_type(8)))  float    v8f;

// Input segment descriptor: lets one GEMM kernel synthesize gathered /
// concatenated / one-hot / product inputs without materializing them.
struct Seg {
    const float* p0;
    const float* p1;
    const int*   rmap;
    int ld;
    int cols;
    int mode;   // 0: p0[rmap?rmap[r]:r][k]      (global, optional row gather)
                // 1: p0[r>>3][k]                (per-set broadcast, z_rep)
                // 2: one_hot(r&7)[k]            (pos of sorted element)
                // 3: one_hot(r)[k]              (identity rows: key tables)
                // 4: one_hot(8)[k]              (pos_n, n==8 always)
                // 5: p0[r/17][k] * p1[r%17][k]  (zp = s ⊙ keys, decoder)
};

__device__ __forceinline__ float fetchSeg(const Seg& s, int r, int kk) {
    switch (s.mode) {
        case 0: { int rm = s.rmap ? s.rmap[r] : r;
                  return s.p0[(size_t)rm * s.ld + kk]; }
        case 1: return s.p0[(size_t)(r >> 3) * s.ld + kk];
        case 2: return ((r & 7) == kk) ? 1.f : 0.f;
        case 3: return (r == kk) ? 1.f : 0.f;
        case 4: return (kk == 8) ? 1.f : 0.f;
        default: return s.p0[(size_t)(r / 17) * s.ld + kk] *
                        s.p1[(size_t)(r % 17) * s.ld + kk];
    }
}

// Pre-pack weights W[K,C] (f32 row-major) into per-lane WMMA B-fragment
// layout, f16, zero-padded: Wp[tile][chunk][lane][e] where
//   col n = tile*16 + (lane&15),  k = chunk*32 + 8*(lane>>4) + e (e<8), chunk*32 + 16 + 8*(lane>>4) + (e-8) (e>=8).
// Each lane then loads its 16 f16 (32 contiguous bytes) with b128 loads.
__global__ void packW_kernel(const float* __restrict__ W, _Float16* __restrict__ Wp,
                             int K, int C) {
    int nkc = ((K + 31) & ~31) >> 5;
    int nct = (C + 15) >> 4;
    int total = nct * nkc * 512;
    int t = blockIdx.x * blockDim.x + threadIdx.x;
    if (t >= total) return;
    int e     = t & 15;
    int lane  = (t >> 4) & 31;
    int blk   = t >> 9;            // tile * nkc + chunk
    int chunk = blk % nkc;
    int tile  = blk / nkc;
    int n = tile * 16 + (lane & 15);
    int k = chunk * 32 + 8 * (lane >> 4) + (e < 8 ? e : 8 + e);
    float w = (k < K && n < C) ? W[(size_t)k * C + n] : 0.f;
    Wp[t] = (_Float16)w;
}

// Software-pipelined per-wave GEMM over NT column tiles (NT compile-time):
// iteration kc computes WMMA with fragments loaded during iteration kc-1,
// then issues loads for kc+1 -> loads overlap XDL work, partial loadcnt waits.
template<int NT>
__device__ __forceinline__ void gemm_tiles(
    const _Float16 (*sX)[MAX_KP], float (*sO)[MAX_CP],
    const _Float16* wl, int wave, int nkc, int m, int h)
{
    int tI[NT];
    #pragma unroll
    for (int j = 0; j < NT; ++j) tI[j] = wave + 8 * j;

    v8f acc[NT];
    #pragma unroll
    for (int j = 0; j < NT; ++j) acc[j] = (v8f){};

    // preload chunk 0
    v8h alo = *(const v8h*)&sX[m][8 * h];
    v8h ahi = *(const v8h*)&sX[m][16 + 8 * h];
    v16h a = __builtin_shufflevector(alo, ahi,
                 0, 1, 2, 3, 4, 5, 6, 7, 8, 9, 10, 11, 12, 13, 14, 15);
    v16h bfr[NT];
    #pragma unroll
    for (int j = 0; j < NT; ++j)
        bfr[j] = *(const v16h*)(wl + ((size_t)(tI[j] * nkc) << 9));

    for (int kc = 0; kc < nkc; ++kc) {
        #pragma unroll
        for (int j = 0; j < NT; ++j)
            acc[j] = __builtin_amdgcn_wmma_f32_16x16x32_f16(
                         false, a, false, bfr[j], (short)0, acc[j], false, false);
        if (kc + 1 < nkc) {                       // uniform (scalar) branch
            const int k0 = (kc + 1) << 5;
            v8h nlo = *(const v8h*)&sX[m][k0 + 8 * h];
            v8h nhi = *(const v8h*)&sX[m][k0 + 16 + 8 * h];
            a = __builtin_shufflevector(nlo, nhi,
                    0, 1, 2, 3, 4, 5, 6, 7, 8, 9, 10, 11, 12, 13, 14, 15);
            #pragma unroll
            for (int j = 0; j < NT; ++j)
                bfr[j] = *(const v16h*)(wl + ((size_t)(tI[j] * nkc + kc + 1) << 9));
        }
    }
    // D layout: VGPR i -> row 8*h + i, col = tile*16 + (lane&15)
    #pragma unroll
    for (int j = 0; j < NT; ++j) {
        #pragma unroll
        for (int i = 0; i < 8; ++i)
            sO[8 * h + i][(tI[j] << 4) + m] = acc[j][i];
    }
}

// Fused layer: Y = act( LN( X @ Wp + b ) * g + be ), X = [segA | segB] rows.
// One block = 16 rows; 8 wave32 sweep col tiles (stride 8, up to 3 tiles/wave).
__global__ __launch_bounds__(256)
void mlp_layer_kernel(Seg sa, Seg sb,
                      const _Float16* __restrict__ Wp, const float* __restrict__ bias,
                      const float* __restrict__ gam, const float* __restrict__ bet,
                      float* __restrict__ Y, int ldy, int R, int C, int act, int ln)
{
    __shared__ _Float16 sX[16][MAX_KP];
    __shared__ float    sO[16][MAX_CP];

    const int K    = sa.cols + sb.cols;
    const int Kpad = (K + 31) & ~31;
    const int tid  = threadIdx.x;
    const int rowBase = blockIdx.x * 16;

    // ---- stage input rows (f32 -> f16) into LDS, zero-padded ----
    for (int t = tid; t < 16 * Kpad; t += 256) {
        int row = t / Kpad;
        int k   = t - row * Kpad;
        int gr  = rowBase + row;
        float v = 0.f;
        if (gr < R && k < K) {
            v = (k < sa.cols) ? fetchSeg(sa, gr, k) : fetchSeg(sb, gr, k - sa.cols);
        }
        sX[row][k] = (_Float16)v;
    }
    __syncthreads();

    // ---- WMMA GEMM (tile-count branch taken once per wave) ----
    const int lane = tid & 31;
    const int wave = tid >> 5;
    const int m    = lane & 15;     // A row / B col within tile
    const int h    = lane >> 4;     // lane half
    const int nct  = (C + 15) >> 4;
    const int nkc  = Kpad >> 5;

    const _Float16* wl = Wp + ((size_t)lane << 4);
    const int rem = nct - wave;
    const int cnt = (rem <= 0) ? 0 : ((rem + 7) >> 3);   // tiles for this wave
    if      (cnt == 3) gemm_tiles<3>(sX, sO, wl, wave, nkc, m, h);
    else if (cnt == 2) gemm_tiles<2>(sX, sO, wl, wave, nkc, m, h);
    else if (cnt == 1) gemm_tiles<1>(sX, sO, wl, wave, nkc, m, h);
    __syncthreads();

    // ---- epilogue: bias + LayerNorm (per row over C) + activation ----
    const int ty = tid >> 4, tx = tid & 15;
    const int gr = rowBase + ty;
    float s1 = 0.f, s2 = 0.f;
    for (int c = tx; c < C; c += 16) {
        float hv = sO[ty][c] + bias[c];
        sO[ty][c] = hv;
        s1 += hv; s2 += hv * hv;
    }
    #pragma unroll
    for (int o = 8; o > 0; o >>= 1) {
        s1 += __shfl_xor(s1, o, 16);
        s2 += __shfl_xor(s2, o, 16);
    }
    const float invC = 1.f / (float)C;
    const float mean = s1 * invC;
    const float var  = s2 * invC - mean * mean;
    const float rstd = rsqrtf(var + 1e-5f);
    if (gr < R) {
        for (int c = tx; c < C; c += 16) {
            float hv = sO[ty][c];
            if (ln) hv = (hv - mean) * rstd * gam[c] + bet[c];
            if      (act == 1) hv = tanhf(hv);
            else if (act == 2) hv = 1.f / (1.f + expf(-hv));
            Y[(size_t)gr * ldy + c] = hv;
        }
    }
}

// ---------------- auxiliary kernels ----------------
__global__ void mag_kernel(const float* __restrict__ x, const float* __restrict__ rw,
                           const float* __restrict__ rb, float* __restrict__ mag, int n) {
    int r = blockIdx.x * blockDim.x + threadIdx.x;
    if (r >= n) return;
    const float* xr = x + (size_t)r * Dd;
    float acc = 0.f;
    for (int k = 0; k < Dd; ++k) acc += xr[k] * rw[k];
    mag[r] = acc + rb[0];
}

// stable ascending rank within each group of 8 (matches lexsort((mag,batch)))
__global__ void sort8_kernel(const float* __restrict__ mag, int* __restrict__ idx, int nsets) {
    int b = blockIdx.x * blockDim.x + threadIdx.x;
    if (b >= nsets) return;
    float m[8];
    #pragma unroll
    for (int j = 0; j < 8; ++j) m[j] = mag[b * 8 + j];
    #pragma unroll
    for (int j = 0; j < 8; ++j) {
        int rank = 0;
        #pragma unroll
        for (int i = 0; i < 8; ++i)
            rank += (m[i] < m[j]) || (m[i] == m[j] && i < j);
        idx[b * 8 + rank] = b * 8 + j;
    }
}

// out[b][c] = sum_j v[(8b+j)][c] * tab[j][c]   (deepset key/value reduce)
__global__ void wsum8_kernel(const float* __restrict__ v, const float* __restrict__ tab,
                             float* __restrict__ out, int nsets, int Cc) {
    int t = blockIdx.x * blockDim.x + threadIdx.x;
    if (t >= nsets * Cc) return;
    int b = t / Cc, c = t - b * Cc;
    float acc = 0.f;
    for (int j = 0; j < 8; ++j)
        acc += v[(size_t)(b * 8 + j) * Cc + c] * tab[(size_t)j * Cc + c];
    out[(size_t)b * Cc + c] = acc;
}

// out[b][c] = sum_j a[(8b+j)][c] * b2[(8b+j)][c]   (main pass reduce)
__global__ void prodsum8_kernel(const float* __restrict__ a, const float* __restrict__ b2,
                                float* __restrict__ out, int nsets, int Cc) {
    int t = blockIdx.x * blockDim.x + threadIdx.x;
    if (t >= nsets * Cc) return;
    int b = t / Cc, c = t - b * Cc;
    float acc = 0.f;
    for (int j = 0; j < 8; ++j)
        acc += a[(size_t)(b * 8 + j) * Cc + c] * b2[(size_t)(b * 8 + j) * Cc + c];
    out[(size_t)b * Cc + c] = acc;
}

__global__ void rsample_kernel(const float* __restrict__ mu, const float* __restrict__ sg,
                               const float* __restrict__ eps, float* __restrict__ s, int n) {
    int t = blockIdx.x * blockDim.x + threadIdx.x;
    if (t < n) s[t] = mu[t] + fabsf(sg[t]) * eps[t];
}

__global__ void argmax_kernel(const float* __restrict__ logits, float* __restrict__ np_f,
                              float* __restrict__ mask, int* __restrict__ np_i, int nsets) {
    int b = blockIdx.x * blockDim.x + threadIdx.x;
    if (b >= nsets) return;
    const float* L = logits + (size_t)b * Mm;
    int best = 0; float bv = L[0];
    for (int mm = 1; mm < Mm; ++mm) { float v = L[mm]; if (v > bv) { bv = v; best = mm; } }
    np_i[b] = best;
    np_f[b] = (float)best;
    for (int mm = 0; mm < Mm; ++mm) mask[(size_t)b * Mm + mm] = (mm < best) ? 1.f : 0.f;
}

__global__ void maskmul_kernel(float* __restrict__ xr, const int* __restrict__ np_i, int total) {
    int t = blockIdx.x * blockDim.x + threadIdx.x;
    if (t >= total) return;
    int r = t >> 7;               // row in [0, B*M)
    int b = r / Mm, mm = r - b * Mm;
    if (mm >= np_i[b]) xr[t] = 0.f;
}

// ---------------- host-side helpers ----------------
static inline Seg segG(const float* p, int ld, int cols, const int* rmap = nullptr) {
    Seg s; s.p0 = p; s.p1 = nullptr; s.rmap = rmap; s.ld = ld; s.cols = cols; s.mode = 0; return s;
}
static inline Seg segSet(const float* p, int ld, int cols) {
    Seg s; s.p0 = p; s.p1 = nullptr; s.rmap = nullptr; s.ld = ld; s.cols = cols; s.mode = 1; return s;
}
static inline Seg segMode(int mode, int cols) {
    Seg s; s.p0 = nullptr; s.p1 = nullptr; s.rmap = nullptr; s.ld = 0; s.cols = cols; s.mode = mode; return s;
}
static inline Seg segProd(const float* p0, const float* p1, int ld, int cols) {
    Seg s; s.p0 = p0; s.p1 = p1; s.rmap = nullptr; s.ld = ld; s.cols = cols; s.mode = 5; return s;
}
static inline Seg segNone() { return segMode(2, 0); }

// pack W into wpack, then run the fused layer
static inline void mlp(hipStream_t st, _Float16* wpack, Seg a, Seg b,
                       const float* W, const float* bias,
                       const float* g, const float* be, float* Y, int ldy,
                       int R, int C, int act, int ln) {
    int K = a.cols + b.cols;
    int nkc = ((K + 31) & ~31) >> 5;
    int nct = (C + 15) >> 4;
    int ptotal = nct * nkc * 512;
    packW_kernel<<<(ptotal + 255) / 256, 256, 0, st>>>(W, wpack, K, C);
    int grid = (R + 15) / 16;
    mlp_layer_kernel<<<grid, 256, 0, st>>>(a, b, wpack, bias, g, be, Y, ldy, R, C, act, ln);
}

extern "C" void kernel_launch(void* const* d_in, const int* in_sizes, int n_in,
                              void* d_out, int out_size, void* d_ws, size_t ws_size,
                              hipStream_t stream) {
    (void)in_sizes; (void)n_in; (void)out_size; (void)ws_size;
    const float* x   = (const float*)d_in[0];
    const float* eps = (const float*)d_in[2];
    auto P = [&](int i) { return (const float*)d_in[i]; };
    // param flat order (W,b[,g,be] per layer):
    // 3..8  key_ds   | 9..14 val_ds | 15..18 enc_ds | 19..24 key_main
    // 25..30 val_main| 31..34 enc_main | 35..40 mu | 41..46 sigma
    // 47 rank_w 48 rank_b | 49..54 key_dec | 55..58 dec | 59..64 size_pred

    // ---- workspace carve ----
    char* wsb = (char*)d_ws;
    size_t off = 0;
    auto alloc = [&](size_t bytes) -> void* {
        void* p = wsb + off; off += (bytes + 255) & ~(size_t)255; return p;
    };
    float*     mag    = (float*)alloc((size_t)Nn * 4);
    int*       idx    = (int*)  alloc((size_t)Nn * 4);
    _Float16*  wpack  = (_Float16*)alloc((size_t)MAX_KP * MAX_CP * 2);
    float*     ktmp   = (float*)alloc(16 * 72 * 4);
    float*     ktab   = (float*)alloc(16 * 128 * 4);
    float*     kdtmp  = (float*)alloc(32 * 136 * 4);
    float*     keys   = (float*)alloc(32 * 256 * 4);
    float*     zds    = (float*)alloc((size_t)Bb * 128 * 4);
    float*     y2ds   = (float*)alloc((size_t)Bb * 128 * 4);
    float*     y2     = (float*)alloc((size_t)Bb * 256 * 4);
    float*     zz     = (float*)alloc((size_t)Bb * 256 * 4);
    float*     muv    = (float*)alloc((size_t)Bb * 256 * 4);
    float*     sig    = (float*)alloc((size_t)Bb * 256 * 4);
    float*     sv     = (float*)alloc((size_t)Bb * 256 * 4);
    float*     hsm    = (float*)alloc((size_t)Bb * 264 * 4);
    int*       npredi = (int*)  alloc((size_t)Bb * 4);
    float*     bufA   = (float*)alloc((size_t)Nn * 256 * 4);
    float*     bufB   = (float*)alloc((size_t)Nn * 256 * 4);
    float*     bufC   = (float*)alloc((size_t)Nn * 256 * 4);
    float*     hd     = bufB;   // dec hidden [R2,192] reuses bufB+bufC region

    // ---- output carve: (xr, mask, n_pred, n_logits) ----
    float* out    = (float*)d_out;
    float* o_xr   = out;
    float* o_mask = o_xr + (size_t)Bb * Mm * Dd;
    float* o_np   = o_mask + (size_t)Bb * Mm;
    float* o_nl   = o_np + Bb;

    const int TANH = 1, SIGM = 2, NONE = 0;

    // ---- rank + per-set sort ----
    mag_kernel  <<<(Nn + 255) / 256, 256, 0, stream>>>(x, P(47), P(48), mag, Nn);
    sort8_kernel<<<(Bb + 255) / 256, 256, 0, stream>>>(mag, idx, Bb);

    // ---- key_ds table: MLP on one-hot rows k=0..7 -> [8,128] ----
    mlp(stream, wpack, segMode(3, Mm), segNone(), P(3), P(4), P(5), P(6), ktmp, 72, 8, 72, TANH, 1);
    mlp(stream, wpack, segG(ktmp, 72, 72), segNone(), P(7), P(8), nullptr, nullptr, ktab, 128, 8, 128, NONE, 0);

    // ---- val_ds on sorted x ----
    mlp(stream, wpack, segG(x, Dd, Dd, idx), segNone(), P(9), P(10), P(11), P(12), bufA, 128, Nn, 128, TANH, 1);
    mlp(stream, wpack, segG(bufA, 128, 128), segNone(), P(13), P(14), nullptr, nullptr, bufB, 128, Nn, 128, NONE, 0);
    wsum8_kernel<<<(Bb * 128 + 255) / 256, 256, 0, stream>>>(bufB, ktab, y2ds, Bb, 128);

    // ---- enc_ds (no LN) ----
    mlp(stream, wpack, segG(y2ds, 128, 128), segNone(), P(15), P(16), nullptr, nullptr, hsm, 128, Bb, 128, TANH, 0);
    mlp(stream, wpack, segG(hsm, 128, 128), segNone(), P(17), P(18), nullptr, nullptr, zds, 128, Bb, 128, NONE, 0);

    // ---- val_main on [xs | z_rep] ----
    mlp(stream, wpack, segG(x, Dd, Dd, idx), segSet(zds, 128, 128), P(25), P(26), P(27), P(28), bufA, 256, Nn, 256, TANH, 1);
    mlp(stream, wpack, segG(bufA, 256, 256), segNone(), P(29), P(30), nullptr, nullptr, bufB, 256, Nn, 256, NONE, 0);

    // ---- key_main on [one_hot(rank) | z_rep] ----
    mlp(stream, wpack, segMode(2, Mm), segSet(zds, 128, 128), P(19), P(20), P(21), P(22), bufA, 200, Nn, 200, TANH, 1);
    mlp(stream, wpack, segG(bufA, 200, 200), segNone(), P(23), P(24), nullptr, nullptr, bufC, 256, Nn, 256, NONE, 0);
    prodsum8_kernel<<<(Bb * 256 + 255) / 256, 256, 0, stream>>>(bufB, bufC, y2, Bb, 256);

    // ---- enc_main on [y2 | one_hot(n=8)] (no LN) ----
    mlp(stream, wpack, segG(y2, 256, 256), segMode(4, Mm), P(31), P(32), nullptr, nullptr, hsm, 264, Bb, 264, TANH, 0);
    mlp(stream, wpack, segG(hsm, 264, 264), segNone(), P(33), P(34), nullptr, nullptr, zz, 256, Bb, 256, NONE, 0);

    // ---- mu / sigma / rsample ----
    mlp(stream, wpack, segG(zz, 256, 256), segNone(), P(35), P(36), P(37), P(38), hsm, 256, Bb, 256, TANH, 1);
    mlp(stream, wpack, segG(hsm, 256, 256), segNone(), P(39), P(40), nullptr, nullptr, muv, 256, Bb, 256, NONE, 0);
    mlp(stream, wpack, segG(zz, 256, 256), segNone(), P(41), P(42), P(43), P(44), hsm, 256, Bb, 256, SIGM, 1);
    mlp(stream, wpack, segG(hsm, 256, 256), segNone(), P(45), P(46), nullptr, nullptr, sig, 256, Bb, 256, NONE, 0);
    rsample_kernel<<<(Bb * 256 + 255) / 256, 256, 0, stream>>>(muv, sig, eps, sv, Bb * 256);

    // ---- size_pred -> n_logits, argmax -> n_pred, mask ----
    mlp(stream, wpack, segG(sv, 256, 256), segNone(), P(59), P(60), P(61), P(62), hsm, 136, Bb, 136, TANH, 1);
    mlp(stream, wpack, segG(hsm, 136, 136), segNone(), P(63), P(64), nullptr, nullptr, o_nl, Mm, Bb, Mm, NONE, 0);
    argmax_kernel<<<(Bb + 255) / 256, 256, 0, stream>>>(o_nl, o_np, o_mask, npredi, Bb);

    // ---- key_dec table on I_17 -> keys [17,256] ----
    mlp(stream, wpack, segMode(3, Mm), segNone(), P(49), P(50), P(51), P(52), kdtmp, 136, Mm, 136, TANH, 1);
    mlp(stream, wpack, segG(kdtmp, 136, 136), segNone(), P(53), P(54), nullptr, nullptr, keys, 256, Mm, 256, NONE, 0);

    // ---- decoder on zp = s[b] ⊙ keys[m], rows = B*M ----
    mlp(stream, wpack, segProd(sv, keys, 256, 256), segNone(), P(55), P(56), nullptr, nullptr, hd, 192, R2, 192, TANH, 0);
    mlp(stream, wpack, segG(hd, 192, 192), segNone(), P(57), P(58), nullptr, nullptr, o_xr, Dd, R2, Dd, NONE, 0);
    maskmul_kernel<<<(R2 * Dd + 255) / 256, 256, 0, stream>>>(o_xr, npredi, R2 * Dd);
}
